// BiLSTM_CRF_14525579395223
// MI455X (gfx1250) — matrix operand order repaired and checked by reading, not compiled, not run on hardware
//
#include <hip/hip_runtime.h>
#include <hip/hip_bf16.h>
#include <math.h>

// ---------------------------------------------------------------------------
// BiLSTM-CRF for MI455X (gfx1250, wave32, WMMA).
// GEMMs use v_wmma_f32_16x16x32_f16 (f32 accumulate). LSTM recurrence is a
// persistent workgroup per direction: h/c live in LDS, W_hh (f16, 512KB/dir)
// stays L2-resident, 16 waves each own a 16-wide hidden slice (all 4 gates).
// ---------------------------------------------------------------------------

typedef __attribute__((ext_vector_type(16))) _Float16 v16h;
typedef __attribute__((ext_vector_type(8)))  float    v8f;

union U16H { v16h v; _Float16 h[16]; uint4 q[2]; };

#define SEQ      512
#define BATCH    64
#define EMB      256
#define HID      256          // per-direction hidden
#define G4       1024         // 4*HID (gates i,f,g,o)
#define KTAG     12
#define START_TAG 10
#define STOP_TAG  11
#define NEGV     (-10000.0f)

// ---- workspace layout (bytes) ----
#define WS_WFIH16  ((size_t)0)                       // 1024*256 f16 = 512KB
#define WS_WBIH16  (WS_WFIH16 + (size_t)G4*EMB*2)
#define WS_WFHH16  (WS_WBIH16 + (size_t)G4*EMB*2)
#define WS_WBHH16  (WS_WFHH16 + (size_t)G4*HID*2)
#define WS_XPROJF  (WS_WBHH16 + (size_t)G4*HID*2)    // [T][B][4H] f16 = 64MB
#define WS_XPROJB  (WS_XPROJF + (size_t)SEQ*BATCH*G4*2)
#define WS_HF      (WS_XPROJB + (size_t)SEQ*BATCH*G4*2)   // [T][B][H] f32
#define WS_HB      (WS_HF     + (size_t)SEQ*BATCH*HID*4)
#define WS_FEATS   (WS_HB     + (size_t)SEQ*BATCH*HID*4)  // [T][B][K] f32
#define WS_BPTR    (WS_FEATS  + (size_t)SEQ*BATCH*KTAG*4) // [T][B][K] u8

// ---------------------------------------------------------------------------
// Kernel 1: convert the four weight matrices (each 1024x256 f32) to f16.
// ---------------------------------------------------------------------------
__global__ void cvt_weights_kernel(const float* __restrict__ a,
                                   const float* __restrict__ b,
                                   const float* __restrict__ c,
                                   const float* __restrict__ d,
                                   _Float16* oa, _Float16* ob,
                                   _Float16* oc, _Float16* od) {
    int idx   = blockIdx.x * blockDim.x + threadIdx.x;   // 0..262143
    int which = blockIdx.y;
    const float* s = (which == 0) ? a : (which == 1) ? b : (which == 2) ? c : d;
    _Float16*    o = (which == 0) ? oa : (which == 1) ? ob : (which == 2) ? oc : od;
    o[idx] = (_Float16)s[idx];
}

// ---------------------------------------------------------------------------
// Kernel 2: embedding gather + input projection (both biases folded in).
//   xproj[t][b][g] = f16( emb(sentence[b][t]) . Wih[g,:] + b_ih[g] + b_hh[g] )
// Tile: 16 rows (one t, 16 b's) staged as f16 in LDS; each of 8 waves does one
// 16-wide N tile with 8 WMMA k-steps (K = 256).
// ---------------------------------------------------------------------------
__global__ void embed_proj_kernel(const int*   __restrict__ sentence,
                                  const float* __restrict__ embed_table,
                                  const _Float16* __restrict__ WihF,
                                  const _Float16* __restrict__ WihB,
                                  const float* __restrict__ bfih,
                                  const float* __restrict__ bfhh,
                                  const float* __restrict__ bbih,
                                  const float* __restrict__ bbhh,
                                  _Float16* __restrict__ xprojF,
                                  _Float16* __restrict__ xprojB) {
    __shared__ _Float16 Atile[16 * EMB];   // 8KB

    const int dir = blockIdx.z;
    const _Float16* W16 = dir ? WihB : WihF;
    const float*    bi  = dir ? bbih : bfih;
    const float*    bh  = dir ? bbhh : bfhh;
    _Float16*       xo  = dir ? xprojB : xprojF;

    const int mtile = blockIdx.x;          // 0..2047
    const int t  = mtile >> 2;             // row = t*64 + b
    const int b0 = (mtile & 3) * 16;

    const int tid = threadIdx.x;           // 256 threads
    {   // cooperative gather + f32->f16 convert: 16 rows x 256 cols
        const int row = tid >> 4;
        const int c0  = (tid & 15) * 16;
        const int widx = sentence[(b0 + row) * SEQ + t];
        const float* src = embed_table + (size_t)widx * EMB + c0;
        #pragma unroll
        for (int i = 0; i < 16; ++i)
            Atile[row * EMB + c0 + i] = (_Float16)src[i];
    }
    __syncthreads();

    const int wave = tid >> 5, lane = tid & 31;
    const int col  = lane & 15, half = lane >> 4;
    const int n0   = blockIdx.y * 128 + wave * 16;   // N tile base in [0,1024)

    v8f acc = {};
    for (int k0 = 0; k0 < EMB; k0 += 32) {
        U16H a, bm;
        // A layout (16-bit 16x32): lane<16 holds K{k0..k0+7, k0+16..k0+23}
        const _Float16* ap = &Atile[(lane & 15) * EMB + k0 + half * 8];
        a.q[0] = *(const uint4*)ap;
        a.q[1] = *(const uint4*)(ap + 16);
        // B layout (32x16): lane-column n0+col, K{k0+8h..+7, k0+16+8h..+7}
        const _Float16* bp = W16 + (size_t)(n0 + col) * EMB + k0 + half * 8;
        bm.q[0] = *(const uint4*)bp;
        bm.q[1] = *(const uint4*)(bp + 16);
        acc = __builtin_amdgcn_wmma_f32_16x16x32_f16(
                  false, a.v, false, bm.v, (short)0, acc, false, false);
    }

    const int g = n0 + col;
    const float bias = bi[g] + bh[g];
    #pragma unroll
    for (int r = 0; r < 8; ++r) {
        const int b = b0 + r + half * 8;   // C layout: M = r + 8*(lane>=16)
        xo[((size_t)(t * BATCH + b)) * G4 + g] = (_Float16)(acc[r] + bias);
    }
}

// ---------------------------------------------------------------------------
// Kernel 3: persistent LSTM scan, one workgroup (512 thr = 16 waves) per
// direction. LDS: h16 [64][256] f16 (A operand), c [64][256] f32.
// Wave w owns hidden slice j0=w*16: computes gate tiles (i,f,g,o) for that
// slice over all 4 M-tiles -> 128 WMMA / step / wave. Two barriers per step.
// ---------------------------------------------------------------------------
__global__ void lstm_scan_kernel(const _Float16* __restrict__ WhhF,
                                 const _Float16* __restrict__ WhhB,
                                 const _Float16* __restrict__ xprojF,
                                 const _Float16* __restrict__ xprojB,
                                 float* __restrict__ houtF,
                                 float* __restrict__ houtB) {
    extern __shared__ char smem[];
    _Float16* h16 = (_Float16*)smem;                     // 32KB
    float*    cst = (float*)(smem + BATCH * HID * 2);    // 64KB

    const int dir = blockIdx.x;
    const _Float16* Whh = dir ? WhhB : WhhF;
    const _Float16* xp  = dir ? xprojB : xprojF;
    float*          ho  = dir ? houtB : houtF;

    const int tid = threadIdx.x;
    for (int i = tid; i < BATCH * HID; i += blockDim.x) {
        h16[i] = (_Float16)0.f;
        cst[i] = 0.f;
    }
    __syncthreads();

    const int wave = tid >> 5, lane = tid & 31;
    const int col  = lane & 15, half = lane >> 4;
    const int j0   = wave * 16;
    const int j    = j0 + col;               // this lane's hidden column

    for (int step = 0; step < SEQ; ++step) {
        const int t = dir ? (SEQ - 1 - step) : step;
        float hreg[4][8];

        #pragma unroll
        for (int mt = 0; mt < 4; ++mt) {
            v8f ai = {}, af = {}, ag = {}, ao = {};
            for (int k0 = 0; k0 < HID; k0 += 32) {
                U16H a;
                const _Float16* ap =
                    &h16[(mt * 16 + (lane & 15)) * HID + k0 + half * 8];
                a.q[0] = *(const uint4*)ap;
                a.q[1] = *(const uint4*)(ap + 16);

                const _Float16* bp = Whh + (size_t)j * HID + k0 + half * 8;
                if (k0 + 32 < HID) __builtin_prefetch(bp + 32, 0, 1);
                U16H b0v, b1v, b2v, b3v;   // gate rows g*256 + j
                b0v.q[0] = *(const uint4*)(bp);
                b0v.q[1] = *(const uint4*)(bp + 16);
                b1v.q[0] = *(const uint4*)(bp + 256 * HID);
                b1v.q[1] = *(const uint4*)(bp + 256 * HID + 16);
                b2v.q[0] = *(const uint4*)(bp + 512 * HID);
                b2v.q[1] = *(const uint4*)(bp + 512 * HID + 16);
                b3v.q[0] = *(const uint4*)(bp + 768 * HID);
                b3v.q[1] = *(const uint4*)(bp + 768 * HID + 16);

                ai = __builtin_amdgcn_wmma_f32_16x16x32_f16(
                         false, a.v, false, b0v.v, (short)0, ai, false, false);
                af = __builtin_amdgcn_wmma_f32_16x16x32_f16(
                         false, a.v, false, b1v.v, (short)0, af, false, false);
                ag = __builtin_amdgcn_wmma_f32_16x16x32_f16(
                         false, a.v, false, b2v.v, (short)0, ag, false, false);
                ao = __builtin_amdgcn_wmma_f32_16x16x32_f16(
                         false, a.v, false, b3v.v, (short)0, ao, false, false);
            }
            // epilogue: add x-projection, activations, c/h update
            #pragma unroll
            for (int r = 0; r < 8; ++r) {
                const int b = mt * 16 + r + half * 8;
                const size_t xb = ((size_t)(t * BATCH + b)) * G4 + j;
                float gi = ai[r] + (float)xp[xb];
                float gf = af[r] + (float)xp[xb + 256];
                float gg = ag[r] + (float)xp[xb + 512];
                float go = ao[r] + (float)xp[xb + 768];
                gi = 1.f / (1.f + __expf(-gi));
                gf = 1.f / (1.f + __expf(-gf));
                gg = tanhf(gg);
                go = 1.f / (1.f + __expf(-go));
                const float cv = gf * cst[b * HID + j] + gi * gg;
                cst[b * HID + j] = cv;
                const float hv = go * tanhf(cv);
                hreg[mt][r] = hv;
                ho[((size_t)(t * BATCH + b)) * HID + j] = hv;
            }
        }
        __syncthreads();   // all waves done reading h16 for this step
        #pragma unroll
        for (int mt = 0; mt < 4; ++mt)
            #pragma unroll
            for (int r = 0; r < 8; ++r) {
                const int b = mt * 16 + r + half * 8;
                h16[b * HID + j] = (_Float16)hreg[mt][r];
            }
        __syncthreads();   // new h visible before next step's reads
    }
}

// ---------------------------------------------------------------------------
// Kernel 4: feats[t][b][k] = [hF|hB] . W_out[k,:] + b_out[k]   (K=12: VALU)
// ---------------------------------------------------------------------------
__global__ void feats_kernel(const float* __restrict__ hF,
                             const float* __restrict__ hB,
                             const float* __restrict__ W_out,
                             const float* __restrict__ b_out,
                             float* __restrict__ feats) {
    const int idx = blockIdx.x * blockDim.x + threadIdx.x;
    if (idx >= SEQ * BATCH * KTAG) return;
    const int k  = idx % KTAG;
    const int tb = idx / KTAG;
    const float* wf = W_out + k * (2 * HID);
    const float* hf = hF + (size_t)tb * HID;
    const float* hb = hB + (size_t)tb * HID;
    float s = b_out[k];
    for (int i = 0; i < HID; i += 4) {
        s += hf[i]   * wf[i]   + hf[i+1] * wf[i+1]
           + hf[i+2] * wf[i+2] + hf[i+3] * wf[i+3];
    }
    for (int i = 0; i < HID; i += 4) {
        s += hb[i]   * wf[HID+i]   + hb[i+1] * wf[HID+i+1]
           + hb[i+2] * wf[HID+i+2] + hb[i+3] * wf[HID+i+3];
    }
    feats[idx] = s;
}

// ---------------------------------------------------------------------------
// Kernel 5: Viterbi DP + backtrace. One thread per batch element (K=12).
// ---------------------------------------------------------------------------
__global__ void viterbi_kernel(const int*   __restrict__ slen,
                               const float* __restrict__ feats,
                               const float* __restrict__ trans,
                               unsigned char* __restrict__ bptr,
                               float* __restrict__ out) {
    __shared__ float tr[KTAG * KTAG];
    const int tid = threadIdx.x;   // 64 threads
    if (tid < KTAG * KTAG) tr[tid] = trans[tid];
    __syncthreads();

    const int b = tid;
    float dp[KTAG];
    #pragma unroll
    for (int k = 0; k < KTAG; ++k) dp[k] = NEGV;
    dp[START_TAG] = 0.f;
    const int len = slen[b];

    for (int t = 0; t < SEQ; ++t) {
        const float* f = feats + ((size_t)t * BATCH + b) * KTAG;
        const bool m = (t < len);
        float ndp[KTAG];
        unsigned char bp[KTAG];
        #pragma unroll
        for (int k = 0; k < KTAG; ++k) {
            float best = dp[0] + tr[k * KTAG];
            int   bi   = 0;
            #pragma unroll
            for (int p = 1; p < KTAG; ++p) {
                const float s = dp[p] + tr[k * KTAG + p];
                if (s > best) { best = s; bi = p; }   // first-max tie-break
            }
            ndp[k] = best + f[k];
            bp[k]  = (unsigned char)bi;
        }
        unsigned char* bpo = bptr + ((size_t)t * BATCH + b) * KTAG;
        #pragma unroll
        for (int k = 0; k < KTAG; ++k) {
            if (m) dp[k] = ndp[k];
            bpo[k] = m ? bp[k] : (unsigned char)k;
        }
    }

    float best = dp[0] + tr[STOP_TAG * KTAG];
    int   bt   = 0;
    #pragma unroll
    for (int k = 1; k < KTAG; ++k) {
        const float s = dp[k] + tr[STOP_TAG * KTAG + k];
        if (s > best) { best = s; bt = k; }
    }
    out[b] = best;                               // path_scores
    int tag = bt;
    for (int t = SEQ - 1; t >= 0; --t) {         // path[b][t]
        out[BATCH + (size_t)b * SEQ + t] = (float)tag;
        tag = bptr[((size_t)t * BATCH + b) * KTAG + tag];
    }
}

// ---------------------------------------------------------------------------
extern "C" void kernel_launch(void* const* d_in, const int* in_sizes, int n_in,
                              void* d_out, int out_size, void* d_ws, size_t ws_size,
                              hipStream_t stream) {
    (void)in_sizes; (void)n_in; (void)out_size; (void)ws_size;

    const int*   sentence = (const int*)d_in[0];
    const int*   slen     = (const int*)d_in[1];
    const float* embed    = (const float*)d_in[2];
    const float* Wf_ih    = (const float*)d_in[3];
    const float* Wf_hh    = (const float*)d_in[4];
    const float* bf_ih    = (const float*)d_in[5];
    const float* bf_hh    = (const float*)d_in[6];
    const float* Wb_ih    = (const float*)d_in[7];
    const float* Wb_hh    = (const float*)d_in[8];
    const float* bb_ih    = (const float*)d_in[9];
    const float* bb_hh    = (const float*)d_in[10];
    const float* W_out    = (const float*)d_in[11];
    const float* b_out    = (const float*)d_in[12];
    const float* trans    = (const float*)d_in[13];

    char* ws = (char*)d_ws;
    _Float16* WfIh16  = (_Float16*)(ws + WS_WFIH16);
    _Float16* WbIh16  = (_Float16*)(ws + WS_WBIH16);
    _Float16* WfHh16  = (_Float16*)(ws + WS_WFHH16);
    _Float16* WbHh16  = (_Float16*)(ws + WS_WBHH16);
    _Float16* xprojF  = (_Float16*)(ws + WS_XPROJF);
    _Float16* xprojB  = (_Float16*)(ws + WS_XPROJB);
    float*    hF      = (float*)(ws + WS_HF);
    float*    hB      = (float*)(ws + WS_HB);
    float*    feats   = (float*)(ws + WS_FEATS);
    unsigned char* bp = (unsigned char*)(ws + WS_BPTR);
    float*    out     = (float*)d_out;

    // 1) weights -> f16
    cvt_weights_kernel<<<dim3((G4 * EMB) / 256, 4), 256, 0, stream>>>(
        Wf_ih, Wb_ih, Wf_hh, Wb_hh, WfIh16, WbIh16, WfHh16, WbHh16);

    // 2) embedding gather + input projection (WMMA), both directions
    embed_proj_kernel<<<dim3((SEQ * BATCH) / 16, G4 / 128, 2), 256, 0, stream>>>(
        sentence, embed, WfIh16, WbIh16,
        bf_ih, bf_hh, bb_ih, bb_hh, xprojF, xprojB);

    // 3) persistent LSTM scan: 1 WG per direction, 96KB dynamic LDS
    lstm_scan_kernel<<<2, 512, BATCH * HID * 2 + BATCH * HID * 4, stream>>>(
        WfHh16, WbHh16, xprojF, xprojB, hF, hB);

    // 4) tag-space projection
    feats_kernel<<<(SEQ * BATCH * KTAG + 255) / 256, 256, 0, stream>>>(
        hF, hB, W_out, b_out, feats);

    // 5) Viterbi + backtrace -> d_out = [path_scores(64) | path(64*512)]
    viterbi_kernel<<<1, 64, 0, stream>>>(slen, feats, trans, bp, out);
}
